// GraphMambaGMN_15015205666893
// MI455X (gfx1250) — matrix operand order repaired and checked by reading, not compiled, not run on hardware
//
#include <hip/hip_runtime.h>
#include <math.h>

// ---------------------------------------------------------------------------
// Model dimensions (from the reference)
// ---------------------------------------------------------------------------
#define cB   16
#define cN   256
#define cL   16
#define cHC  256
#define cMH  512
#define cDM  256
#define cDS  16
#define cDC  4
#define cDI  512
#define cDTR 16
#define cNL  2
#define cNG  2
#define cML  2
#define cE   2048
#define cINV_TAU 5.0f

#define TOK_LOCAL (cB * cN * cL)   // 65536
#define NODES     (cB * cN)        // 4096

#define DIV_UP(a, b) (((a) + (b) - 1) / (b))

typedef __attribute__((ext_vector_type(16))) _Float16 v16h;
typedef __attribute__((ext_vector_type(8)))  float    v8f;
typedef __attribute__((ext_vector_type(4)))  unsigned int u32x4;
typedef __attribute__((ext_vector_type(8)))  int i32x8;
typedef __attribute__((ext_vector_type(4)))  int i32x4;

// ---------------------------------------------------------------------------
// f32 -> f16 conversion (weights are converted once per launch)
// ---------------------------------------------------------------------------
__global__ void cvt_f16_kernel(const float* __restrict__ s, _Float16* __restrict__ d, int n)
{
  int idx = (int)(blockIdx.x * (unsigned)blockDim.x + threadIdx.x);
  if (idx < n) d[idx] = (_Float16)s[idx];
}

// ---------------------------------------------------------------------------
// Block-cooperative WMMA GEMM: Y[M,N] = epi( X[M,K] @ W[N,K]^T + bias )
//   - block = 256 threads = 8 waves, block tile 64(M) x 32(N)
//   - the 32 x K f16 weight strip is DMA'd into LDS by the Tensor Data Mover
//     (wave 0 issues TENSOR_LOAD_TO_LDS, waits tensorcnt, block barriers)
//   - each wave computes one 16x16 tile with v_wmma_f32_16x16x32_f16
// Requirements: M % 64 == 0, N % 32 == 0, K % 32 == 0, K <= 512
// EPI: 0 = bias, 1 = bias+relu, 2 = 0.5*(aux + acc)  (no bias)
// ---------------------------------------------------------------------------
template<int EPI>
__global__ __launch_bounds__(256) void gemm_wmma_tdm_kernel(
    const float* __restrict__ X, const _Float16* __restrict__ Wh,
    const float* __restrict__ bias, const float* __restrict__ aux,
    float* __restrict__ Y, int M, int N, int K)
{
  __shared__ __align__(16) _Float16 Bs[32 * 512];   // 32 KB weight strip

  int nbn  = N >> 5;
  int bm   = (int)blockIdx.x / nbn;
  int bn   = (int)blockIdx.x - bm * nbn;
  int row0 = bm << 6;
  int col0 = bn << 5;

  const _Float16* gtile = Wh + (size_t)col0 * K;    // 32 contiguous rows of K halves

#if defined(__has_builtin) && __has_builtin(__builtin_amdgcn_tensor_load_to_lds)
  if (threadIdx.x == 0) {
    unsigned long long ga = (unsigned long long)(size_t)gtile;
    unsigned int lds = (unsigned int)(size_t)&Bs[0];          // addr[31:0] = LDS offset
    unsigned int Ku = (unsigned int)K, R = 32u;
    u32x4 g0;
    g0[0] = 1u;                                               // count=1, user desc
    g0[1] = lds;                                              // lds_addr
    g0[2] = (unsigned int)ga;                                 // global_addr[31:0]
    g0[3] = ((unsigned int)(ga >> 32) & 0x01FFFFFFu) | (2u << 30); // addr[56:32] | type=2
    i32x8 g1;
    g1[0] = (int)(1u << 16);                                  // data_size = 2 bytes
    g1[1] = (int)((Ku & 0xFFFFu) << 16);                      // tensor_dim0[15:0]
    g1[2] = (int)((Ku >> 16) | (R << 16));                    // tensor_dim0[31:16] | tensor_dim1[15:0]
    g1[3] = (int)((Ku & 0xFFFFu) << 16);                      // tensor_dim1 hi=0 | tile_dim0=K
    g1[4] = (int)R;                                           // tile_dim1 = 32, tile_dim2 = 0
    g1[5] = (int)Ku;                                          // tensor_dim0_stride = K (elems)
    g1[6] = 0;
    g1[7] = 0;
    i32x4 zz4; zz4[0] = 0; zz4[1] = 0; zz4[2] = 0; zz4[3] = 0;
    i32x8 zz8; zz8[0] = 0; zz8[1] = 0; zz8[2] = 0; zz8[3] = 0;
    zz8[4] = 0; zz8[5] = 0; zz8[6] = 0; zz8[7] = 0;
    __builtin_amdgcn_tensor_load_to_lds(g0, g1, zz4, zz4, zz8, 0);
  }
#if __has_builtin(__builtin_amdgcn_s_wait_tensorcnt)
  if (threadIdx.x < 32) __builtin_amdgcn_s_wait_tensorcnt(0);
#endif
#else
  {
    // Fallback: cooperative copy (tile is contiguous since stride == dim0)
    const uint4* src = (const uint4*)gtile;
    uint4* dst = (uint4*)&Bs[0];
    int n16 = (32 * K) >> 3;                                  // 16B chunks
    for (int i = threadIdx.x; i < n16; i += 256) dst[i] = src[i];
  }
#endif
  __syncthreads();

  int wave  = threadIdx.x >> 5;
  int lane  = threadIdx.x & 31;
  int mw    = wave >> 1;            // 0..3 -> 16-row group
  int nw    = wave & 1;             // 0..1 -> 16-col group
  int lrow  = lane & 15;
  int khalf = lane >> 4;

  const float*    pa_base = X + (size_t)(row0 + (mw << 4) + lrow) * K + khalf * 8;
  const _Float16* pb_base = Bs + (size_t)((nw << 4) + lrow) * K + khalf * 8;

  v8f acc = {};
  for (int k0 = 0; k0 < K; k0 += 32) {
    const float*    pa = pa_base + k0;
    const _Float16* pb = pb_base + k0;
    __builtin_prefetch(pa + 32, 0, 3);                        // global_prefetch_b8
    v16h a, b;
#pragma unroll
    for (int i = 0; i < 8; ++i) { a[i] = (_Float16)pa[i]; a[i + 8] = (_Float16)pa[16 + i]; }
#pragma unroll
    for (int i = 0; i < 8; ++i) { b[i] = pb[i]; b[i + 8] = pb[16 + i]; }
    acc = __builtin_amdgcn_wmma_f32_16x16x32_f16(
        false, a, false, b, (short)0, acc, false, false);
  }

  int n  = col0 + (nw << 4) + lrow;
  int mb = row0 + (mw << 4) + (khalf << 3);
  float bn_ = (EPI == 2) ? 0.f : (bias ? bias[n] : 0.f);
#pragma unroll
  for (int r = 0; r < 8; ++r) {
    float v = acc[r] + bn_;
    if (EPI == 1) v = fmaxf(v, 0.f);
    if (EPI == 2) v = 0.5f * (v + aux[(size_t)(mb + r) * N + n]);
    Y[(size_t)(mb + r) * N + n] = v;
  }
}

// ---------------------------------------------------------------------------
// Single-wave-per-tile WMMA GEMM (used when N % 32 != 0, e.g. x_proj N=48)
// ---------------------------------------------------------------------------
template<int EPI>
__global__ __launch_bounds__(256) void gemm_wmma_kernel(
    const float* __restrict__ X, const _Float16* __restrict__ Wh,
    const float* __restrict__ bias, const float* __restrict__ aux,
    float* __restrict__ Y, int M, int N, int K)
{
  int wave = (int)((blockIdx.x * (unsigned)blockDim.x + threadIdx.x) >> 5);
  int lane = threadIdx.x & 31;
  int ntn  = N >> 4;
  int tm = wave / ntn;
  int tn = wave - tm * ntn;
  if (tm * 16 >= M) return;                 // wave-uniform guard; EXEC stays full
  int row = tm << 4, col = tn << 4;
  int lrow  = lane & 15;
  int khalf = lane >> 4;

  const float*    pa_base = X  + (size_t)(row + lrow) * K + khalf * 8;
  const _Float16* pb_base = Wh + (size_t)(col + lrow) * K + khalf * 8;

  v8f acc = {};
  for (int k0 = 0; k0 < K; k0 += 32) {
    const float*    pa = pa_base + k0;
    const _Float16* pb = pb_base + k0;
    v16h a, b;
#pragma unroll
    for (int i = 0; i < 8; ++i) { a[i] = (_Float16)pa[i]; a[i + 8] = (_Float16)pa[16 + i]; }
#pragma unroll
    for (int i = 0; i < 8; ++i) { b[i] = pb[i]; b[i + 8] = pb[16 + i]; }
    acc = __builtin_amdgcn_wmma_f32_16x16x32_f16(
        false, a, false, b, (short)0, acc, false, false);
  }

  int n  = col + lrow;
  int mb = row + (khalf << 3);
  float bn_ = (EPI == 2) ? 0.f : (bias ? bias[n] : 0.f);
#pragma unroll
  for (int r = 0; r < 8; ++r) {
    float v = acc[r] + bn_;
    if (EPI == 1) v = fmaxf(v, 0.f);
    if (EPI == 2) v = 0.5f * (v + aux[(size_t)(mb + r) * N + n]);
    Y[(size_t)(mb + r) * N + n] = v;
  }
}

// ---------------------------------------------------------------------------
// LayerNorm over DM=256, one wave32 per row
// ---------------------------------------------------------------------------
__global__ void ln_kernel(const float* __restrict__ X, const float* __restrict__ w,
                          const float* __restrict__ b, float* __restrict__ Y, int rows)
{
  int r    = (int)((blockIdx.x * (unsigned)blockDim.x + threadIdx.x) >> 5);
  int lane = threadIdx.x & 31;
  if (r >= rows) return;
  const float* xr = X + (size_t)r * cDM;
  float v[8]; float s = 0.f;
#pragma unroll
  for (int i = 0; i < 8; ++i) { v[i] = xr[lane + i * 32]; s += v[i]; }
#pragma unroll
  for (int o = 16; o > 0; o >>= 1) s += __shfl_xor(s, o, 32);
  float mean = s * (1.f / cDM);
  float vs = 0.f;
#pragma unroll
  for (int i = 0; i < 8; ++i) { float d = v[i] - mean; vs += d * d; }
#pragma unroll
  for (int o = 16; o > 0; o >>= 1) vs += __shfl_xor(vs, o, 32);
  float inv = rsqrtf(vs * (1.f / cDM) + 1e-5f);
  float* yr = Y + (size_t)r * cDM;
#pragma unroll
  for (int i = 0; i < 8; ++i)
    yr[lane + i * 32] = (v[i] - mean) * inv * w[lane + i * 32] + b[lane + i * 32];
}

// ---------------------------------------------------------------------------
// In-place RMSNorm(+relu) over MH=512, one wave32 per row (encoder hidden)
// ---------------------------------------------------------------------------
__global__ void rms_relu_kernel(float* __restrict__ H, const float* __restrict__ w, int rows)
{
  int r    = (int)((blockIdx.x * (unsigned)blockDim.x + threadIdx.x) >> 5);
  int lane = threadIdx.x & 31;
  if (r >= rows) return;
  float* row = H + (size_t)r * cMH;
  float v[16]; float s = 0.f;
#pragma unroll
  for (int i = 0; i < 16; ++i) { v[i] = row[lane + i * 32]; s += v[i] * v[i]; }
#pragma unroll
  for (int o = 16; o > 0; o >>= 1) s += __shfl_xor(s, o, 32);
  float inv = rsqrtf(s * (1.f / cMH) + 1e-6f);
#pragma unroll
  for (int i = 0; i < 16; ++i)
    row[lane + i * 32] = fmaxf(v[i] * inv * w[lane + i * 32], 0.f);
}

// ---------------------------------------------------------------------------
// Depthwise conv (DC=4) + bias + SiLU, direction-aware
// ---------------------------------------------------------------------------
__global__ void conv_silu_kernel(const float* __restrict__ xcr, const float* __restrict__ cw,
                                 const float* __restrict__ cb, float* __restrict__ out,
                                 int tokens, int Lseq, int dir)
{
  int idx = (int)(blockIdx.x * (unsigned)blockDim.x + threadIdx.x);
  if (idx >= tokens * cDI) return;
  int d   = idx & (cDI - 1);
  int tok = idx >> 9;
  int t   = tok % Lseq;
  float acc = cb[d];
#pragma unroll
  for (int j = 0; j < cDC; ++j) {
    int dt = dir ? (3 - j) : (j - 3);
    int tt = t + dt;
    if (tt >= 0 && tt < Lseq)
      acc += cw[d * cDC + j] * xcr[(size_t)(tok + dt) * cDI + d];
  }
  float sg = 1.f / (1.f + __expf(-acc));
  out[(size_t)tok * cDI + d] = acc * sg;
}

// ---------------------------------------------------------------------------
// Fused selective scan (softplus(dt@dt_w^T+dt_b) inline, y=scan+x*D, *silu(z))
// ---------------------------------------------------------------------------
__global__ void scan_kernel(const float* __restrict__ dbc, const float* __restrict__ xc,
                            const float* __restrict__ z, const float* __restrict__ dt_w,
                            const float* __restrict__ dt_b, const float* __restrict__ A_log,
                            const float* __restrict__ Dp, float* __restrict__ g,
                            int nseq, int Lseq, int dir)
{
  int idx = (int)(blockIdx.x * (unsigned)blockDim.x + threadIdx.x);
  if (idx >= nseq * cDI) return;
  int d   = idx & (cDI - 1);
  int seq = idx >> 9;
  float A[cDS], wr[cDS], h[cDS];
#pragma unroll
  for (int s = 0; s < cDS; ++s) {
    A[s]  = -__expf(A_log[d * cDS + s]);
    wr[s] = dt_w[d * cDS + s];
    h[s]  = 0.f;
  }
  float bq = dt_b[d], Dd = Dp[d];
  for (int st = 0; st < Lseq; ++st) {
    int t = dir ? (Lseq - 1 - st) : st;
    size_t tok = (size_t)seq * Lseq + t;
    const float* dr = dbc + tok * 48;
    float dv = bq;
#pragma unroll
    for (int r = 0; r < cDTR; ++r) dv += dr[r] * wr[r];
    float delta = (dv > 20.f) ? dv : log1pf(__expf(dv));
    float xv = xc[tok * cDI + d];
    float dx = delta * xv;
    float acc = 0.f;
#pragma unroll
    for (int s = 0; s < cDS; ++s) {
      h[s] = __expf(delta * A[s]) * h[s] + dx * dr[cDTR + s];
      acc += h[s] * dr[cDTR + cDS + s];
    }
    float yv = acc + xv * Dd;
    float zv = z[tok * cDI + d];
    g[tok * cDI + d] = yv * (zv / (1.f + __expf(-zv)));
  }
}

// ---------------------------------------------------------------------------
// Data movement / elementwise kernels
// ---------------------------------------------------------------------------
__global__ void take_last_kernel(const float* __restrict__ x, float* __restrict__ node,
                                 int nseq, int Lseq)
{
  int idx = (int)(blockIdx.x * (unsigned)blockDim.x + threadIdx.x);
  if (idx >= nseq * cDM) return;
  int s = idx >> 8, c = idx & 255;
  node[idx] = x[((size_t)s * Lseq + (Lseq - 1)) * cDM + c];
}

__global__ void perm_gather_kernel(const float* __restrict__ src, const int* __restrict__ perm,
                                   float* __restrict__ dst)
{
  int idx = (int)(blockIdx.x * (unsigned)blockDim.x + threadIdx.x);
  if (idx >= NODES * cDM) return;
  int bn = idx >> 8, c = idx & 255;
  int b = bn >> 8, i = bn & 255;
  int p = perm[(b << 8) + i];
  dst[idx] = src[(((size_t)(b << 8) + p) << 8) + c];
}

__global__ void perm_scatter_kernel(const float* __restrict__ src, const int* __restrict__ perm,
                                    float* __restrict__ dst)
{
  int idx = (int)(blockIdx.x * (unsigned)blockDim.x + threadIdx.x);
  if (idx >= NODES * cDM) return;
  int bn = idx >> 8, c = idx & 255;
  int b = bn >> 8, i = bn & 255;
  int p = perm[(b << 8) + i];
  dst[(((size_t)(b << 8) + p) << 8) + c] = src[idx];
}

__global__ void edge_gather_kernel(const float* __restrict__ x, const int* __restrict__ edges,
                                   float* __restrict__ Xe)
{
  int idx = (int)(blockIdx.x * (unsigned)blockDim.x + threadIdx.x);
  if (idx >= cB * cE * cDM) return;
  int be = idx >> 8, c = idx & 255;
  int b = be / cE, e = be - b * cE;
  int s = edges[(size_t)(b * 2) * cE + e];
  Xe[idx] = x[(((size_t)b * cN + s) << 8) + c];
}

__global__ void edge_scatter_add_kernel(const float* __restrict__ Msg, const int* __restrict__ edges,
                                        float* __restrict__ m)
{
  int idx = (int)(blockIdx.x * (unsigned)blockDim.x + threadIdx.x);
  if (idx >= cB * cE * cDM) return;
  int be = idx >> 8, c = idx & 255;
  int b = be / cE, e = be - b * cE;
  int dst = edges[(size_t)(b * 2 + 1) * cE + e];
  atomicAdd(&m[(((size_t)b * cN + dst) << 8) + c], Msg[idx]);
}

__global__ void zero_kernel(float* __restrict__ p, int n)
{
  int idx = (int)(blockIdx.x * (unsigned)blockDim.x + threadIdx.x);
  if (idx < n) p[idx] = 0.f;
}

__global__ void copy_kernel(const float* __restrict__ s, float* __restrict__ d, int n)
{
  int idx = (int)(blockIdx.x * (unsigned)blockDim.x + threadIdx.x);
  if (idx < n) d[idx] = s[idx];
}

__global__ void concat2_kernel(const float* __restrict__ a, const float* __restrict__ b,
                               float* __restrict__ cat)
{
  int idx = (int)(blockIdx.x * (unsigned)blockDim.x + threadIdx.x);
  if (idx >= NODES * 2 * cDM) return;
  int t = idx >> 9, c = idx & 511;
  cat[idx] = (c < cDM) ? a[(size_t)t * cDM + c] : b[(size_t)t * cDM + (c - cDM)];
}

__global__ void add_relu_kernel(const float* __restrict__ a, const float* __restrict__ b,
                                float* __restrict__ y, int n)
{
  int idx = (int)(blockIdx.x * (unsigned)blockDim.x + threadIdx.x);
  if (idx < n) y[idx] = a[idx] + fmaxf(b[idx], 0.f);
}

__global__ void row_normalize_kernel(const float* __restrict__ X, float* __restrict__ Y, int rows)
{
  int r    = (int)((blockIdx.x * (unsigned)blockDim.x + threadIdx.x) >> 5);
  int lane = threadIdx.x & 31;
  if (r >= rows) return;
  const float* xr = X + (size_t)r * 256;
  float v[8]; float s = 0.f;
#pragma unroll
  for (int i = 0; i < 8; ++i) { v[i] = xr[lane + i * 32]; s += v[i] * v[i]; }
#pragma unroll
  for (int o = 16; o > 0; o >>= 1) s += __shfl_xor(s, o, 32);
  float inv = 1.f / fmaxf(sqrtf(s), 1e-12f);
  float* yr = Y + (size_t)r * 256;
#pragma unroll
  for (int i = 0; i < 8; ++i) yr[lane + i * 32] = v[i] * inv;
}

// ---------------------------------------------------------------------------
// InfoNCE: one block per row i; masked log-softmax diagonal
// ---------------------------------------------------------------------------
__global__ __launch_bounds__(256) void infonce_rows_kernel(
    const float* __restrict__ Xr, const float* __restrict__ Xc, float* __restrict__ partial)
{
  __shared__ float xi[256];
  __shared__ float red[256];
  __shared__ float diag_s;
  int i = blockIdx.x;
  int tid = threadIdx.x;
  xi[tid] = Xr[(size_t)i * 256 + tid];
  if (tid == 0) diag_s = 0.f;
  __syncthreads();
  int gi = i >> 8;
  float dots[16];
  float mx = -3.4e38f;
#pragma unroll
  for (int k = 0; k < 16; ++k) {
    int j = tid + (k << 8);
    const float* xj = Xc + (size_t)j * 256;
    float dm = 0.f;
    for (int c = 0; c < 256; ++c) dm += xi[c] * xj[c];
    float l = dm * cINV_TAU;
    if (j == i) diag_s = l;
    if (((j >> 8) == gi) && (j != i)) l = -3.4e38f;
    dots[k] = l;
    mx = fmaxf(mx, l);
  }
  red[tid] = mx; __syncthreads();
  for (int s = 128; s > 0; s >>= 1) { if (tid < s) red[tid] = fmaxf(red[tid], red[tid + s]); __syncthreads(); }
  float rmax = red[0]; __syncthreads();
  float se = 0.f;
#pragma unroll
  for (int k = 0; k < 16; ++k)
    if (dots[k] > -3.0e38f) se += __expf(dots[k] - rmax);
  red[tid] = se; __syncthreads();
  for (int s = 128; s > 0; s >>= 1) { if (tid < s) red[tid] += red[tid + s]; __syncthreads(); }
  if (tid == 0) partial[i] = -(diag_s - (rmax + logf(red[0])));
}

__global__ void final_loss_kernel(const float* __restrict__ p1, const float* __restrict__ p2,
                                  float* __restrict__ out)
{
  __shared__ float red[256];
  float s = 0.f;
  for (int i = threadIdx.x; i < NODES; i += 256) s += p1[i] + p2[i];
  red[threadIdx.x] = s; __syncthreads();
  for (int st = 128; st > 0; st >>= 1) { if (threadIdx.x < st) red[threadIdx.x] += red[threadIdx.x + st]; __syncthreads(); }
  if (threadIdx.x == 0) out[0] = 0.5f * red[0] / (float)NODES;
}

// ---------------------------------------------------------------------------
// Host-side orchestration
// ---------------------------------------------------------------------------
struct StackParams {
  const float *A_log, *D, *conv_b, *conv_w, *dt_b, *dt_w, *in_w, *ln_b, *ln_w, *out_w, *xp_w;
  const _Float16 *in_wh, *xp_wh, *out_wh;   // f16 copies (workspace)
};

static void launch_gemm(hipStream_t st, const float* X, const _Float16* Wh, const float* bias,
                        const float* aux, float* Y, int M, int N, int K, int epi)
{
  if ((M & 63) == 0 && (N & 31) == 0 && K <= 512) {
    int blocks = (M >> 6) * (N >> 5);
    switch (epi) {
      case 0: gemm_wmma_tdm_kernel<0><<<blocks, 256, 0, st>>>(X, Wh, bias, aux, Y, M, N, K); break;
      case 1: gemm_wmma_tdm_kernel<1><<<blocks, 256, 0, st>>>(X, Wh, bias, aux, Y, M, N, K); break;
      default: gemm_wmma_tdm_kernel<2><<<blocks, 256, 0, st>>>(X, Wh, bias, aux, Y, M, N, K); break;
    }
  } else {
    long waves = (long)(M / 16) * (N / 16);
    int  grid  = (int)DIV_UP(waves * 32, 256);
    switch (epi) {
      case 0: gemm_wmma_kernel<0><<<grid, 256, 0, st>>>(X, Wh, bias, aux, Y, M, N, K); break;
      case 1: gemm_wmma_kernel<1><<<grid, 256, 0, st>>>(X, Wh, bias, aux, Y, M, N, K); break;
      default: gemm_wmma_kernel<2><<<grid, 256, 0, st>>>(X, Wh, bias, aux, Y, M, N, K); break;
    }
  }
}

static void run_bidir_stack(hipStream_t st, float* x, int tokens, int Lseq, int nlayers,
                            const StackParams& P, float* xn, float* bufA, float* bufB,
                            float* bufC, float* dbc, float* yf)
{
  int nseq = tokens / Lseq;
  for (int l = 0; l < nlayers; ++l) {
    ln_kernel<<<DIV_UP(tokens * 32, 256), 256, 0, st>>>(x, P.ln_w + l * cDM, P.ln_b + l * cDM, xn, tokens);
    for (int d = 0; d < 2; ++d) {
      int ld = l * 2 + d;
      const _Float16* in_wh  = P.in_wh  + (size_t)ld * 2 * cDI * cDM;
      const float*    conv_w = P.conv_w + (size_t)ld * cDI * cDC;
      const float*    conv_b = P.conv_b + (size_t)ld * cDI;
      const _Float16* xp_wh  = P.xp_wh  + (size_t)ld * 48 * cDI;
      const float*    dt_w   = P.dt_w   + (size_t)ld * cDI * cDTR;
      const float*    dt_b   = P.dt_b   + (size_t)ld * cDI;
      const float*    A_log  = P.A_log  + (size_t)ld * cDI * cDS;
      const float*    Dp     = P.D      + (size_t)ld * cDI;
      const _Float16* out_wh = P.out_wh + (size_t)ld * cDM * cDI;

      launch_gemm(st, xn, in_wh,             nullptr, nullptr, bufA, tokens, cDI, cDM, 0);
      launch_gemm(st, xn, in_wh + cDI * cDM, nullptr, nullptr, bufB, tokens, cDI, cDM, 0);
      conv_silu_kernel<<<DIV_UP(tokens * cDI, 256), 256, 0, st>>>(bufA, conv_w, conv_b, bufC, tokens, Lseq, d);
      launch_gemm(st, bufC, xp_wh, nullptr, nullptr, dbc, tokens, 48, cDI, 0);
      scan_kernel<<<DIV_UP(nseq * cDI, 256), 256, 0, st>>>(dbc, bufC, bufB, dt_w, dt_b, A_log, Dp, bufA, nseq, Lseq, d);
      if (d == 0)
        launch_gemm(st, bufA, out_wh, nullptr, nullptr, yf, tokens, cDM, cDI, 0);
      else
        launch_gemm(st, bufA, out_wh, nullptr, yf, x, tokens, cDM, cDI, 2);  // x = 0.5*(yf + yb)
    }
  }
}

struct ViewScratch {
  float *xbuf, *xn, *A, *Bz, *C, *dbc, *yf;
  float *node, *srt, *ng, *mpx, *mpx2, *m, *cat, *rep, *ph;
};

struct HalfWeights {
  const _Float16 *enc_w1, *enc_w2, *mw1, *mw2, *uw1, *uw2, *pw1, *pw2;
};

static void run_view(hipStream_t st, const float* pooled, const int* perm, const int* edges,
                     const float* enc_b1, const float* enc_rms, const float* enc_b2,
                     const StackParams& locl, const StackParams& glob,
                     const float* mb1, const float* mb2, const float* ub1, const float* ub2,
                     const float* pb1, const float* pb2, const HalfWeights& HW,
                     float* zout, ViewScratch& W)
{
  // Encoder: relu(rms(pooled @ w1.T + b1)) @ w2.T + b2
  launch_gemm(st, pooled, HW.enc_w1, enc_b1, nullptr, W.A, TOK_LOCAL, cMH, cHC, 0);
  rms_relu_kernel<<<DIV_UP(TOK_LOCAL * 32, 256), 256, 0, st>>>(W.A, enc_rms, TOK_LOCAL);
  launch_gemm(st, W.A, HW.enc_w2, enc_b2, nullptr, W.xbuf, TOK_LOCAL, cDM, cMH, 0);

  // Local bidirectional stack (seq len L=16)
  run_bidir_stack(st, W.xbuf, TOK_LOCAL, cL, cNL, locl, W.xn, W.A, W.Bz, W.C, W.dbc, W.yf);

  // node = last token; sort; global stack (seq len 256); unsort
  take_last_kernel<<<DIV_UP(NODES * cDM, 256), 256, 0, st>>>(W.xbuf, W.node, NODES, cL);
  perm_gather_kernel<<<DIV_UP(NODES * cDM, 256), 256, 0, st>>>(W.node, perm, W.srt);
  run_bidir_stack(st, W.srt, NODES, cN, cNG, glob, W.xn, W.A, W.Bz, W.C, W.dbc, W.yf);
  perm_scatter_kernel<<<DIV_UP(NODES * cDM, 256), 256, 0, st>>>(W.srt, perm, W.ng);

  // MPNN (2 layers) on ng
  copy_kernel<<<DIV_UP(NODES * cDM, 256), 256, 0, st>>>(W.ng, W.mpx, NODES * cDM);
  float* cur = W.mpx; float* nxt = W.mpx2;
  for (int l = 0; l < cML; ++l) {
    edge_gather_kernel<<<DIV_UP(cB * cE * cDM, 256), 256, 0, st>>>(cur, edges, W.A);
    launch_gemm(st, W.A, HW.mw1 + (size_t)l * cDM * cDM, mb1 + l * cDM, nullptr, W.Bz, cB * cE, cDM, cDM, 1);
    launch_gemm(st, W.Bz, HW.mw2 + (size_t)l * cDM * cDM, mb2 + l * cDM, nullptr, W.C, cB * cE, cDM, cDM, 0);
    zero_kernel<<<DIV_UP(NODES * cDM, 256), 256, 0, st>>>(W.m, NODES * cDM);
    edge_scatter_add_kernel<<<DIV_UP(cB * cE * cDM, 256), 256, 0, st>>>(W.C, edges, W.m);
    concat2_kernel<<<DIV_UP(NODES * 2 * cDM, 256), 256, 0, st>>>(cur, W.m, W.cat);
    launch_gemm(st, W.cat, HW.uw1 + (size_t)l * cDM * 2 * cDM, ub1 + l * cDM, nullptr, W.A, NODES, cDM, 2 * cDM, 1);
    launch_gemm(st, W.A, HW.uw2 + (size_t)l * cDM * cDM, ub2 + l * cDM, nullptr, nxt, NODES, cDM, cDM, 0);
    float* t = cur; cur = nxt; nxt = t;
  }
  add_relu_kernel<<<DIV_UP(NODES * cDM, 256), 256, 0, st>>>(W.ng, cur, W.rep, NODES * cDM);

  // Projection head
  launch_gemm(st, W.rep, HW.pw1, pb1, nullptr, W.ph, NODES, cDM, cDM, 1);
  launch_gemm(st, W.ph, HW.pw2, pb2, nullptr, zout, NODES, 256, cDM, 0);
}

extern "C" void kernel_launch(void* const* d_in, const int* in_sizes, int n_in,
                              void* d_out, int out_size, void* d_ws, size_t ws_size,
                              hipStream_t stream)
{
  (void)in_sizes; (void)n_in; (void)out_size; (void)ws_size;

  const float* pooled1 = (const float*)d_in[0];
  const float* pooled2 = (const float*)d_in[1];
  const int*   perm1   = (const int*)d_in[2];
  const int*   perm2   = (const int*)d_in[3];
  const int*   edges1  = (const int*)d_in[4];
  const int*   edges2  = (const int*)d_in[5];
  // params: pytree (dict keys sorted): enc, glob, local, mpnn, proj
  const float* enc_b1  = (const float*)d_in[6];
  const float* enc_b2  = (const float*)d_in[7];
  const float* enc_rms = (const float*)d_in[8];
  const float* enc_w1  = (const float*)d_in[9];
  const float* enc_w2  = (const float*)d_in[10];
  StackParams glob = {
    (const float*)d_in[11], (const float*)d_in[12], (const float*)d_in[13],
    (const float*)d_in[14], (const float*)d_in[15], (const float*)d_in[16],
    (const float*)d_in[17], (const float*)d_in[18], (const float*)d_in[19],
    (const float*)d_in[20], (const float*)d_in[21], nullptr, nullptr, nullptr };
  StackParams locl = {
    (const float*)d_in[22], (const float*)d_in[23], (const float*)d_in[24],
    (const float*)d_in[25], (const float*)d_in[26], (const float*)d_in[27],
    (const float*)d_in[28], (const float*)d_in[29], (const float*)d_in[30],
    (const float*)d_in[31], (const float*)d_in[32], nullptr, nullptr, nullptr };
  const float* mb1 = (const float*)d_in[33];
  const float* mb2 = (const float*)d_in[34];
  const float* mw1 = (const float*)d_in[35];
  const float* mw2 = (const float*)d_in[36];
  const float* ub1 = (const float*)d_in[37];
  const float* ub2 = (const float*)d_in[38];
  const float* uw1 = (const float*)d_in[39];
  const float* uw2 = (const float*)d_in[40];
  const float* pb1 = (const float*)d_in[41];
  const float* pb2 = (const float*)d_in[42];
  const float* pw1 = (const float*)d_in[43];
  const float* pw2 = (const float*)d_in[44];

  // -------------------- workspace carve-up --------------------
  float* wsf = (float*)d_ws;
  size_t off = 0;
  auto alloc = [&](size_t n) { float* p = wsf + off; off += n; return p; };
  ViewScratch W;
  W.xbuf = alloc((size_t)TOK_LOCAL * cDM);
  W.xn   = alloc((size_t)TOK_LOCAL * cDM);
  W.A    = alloc((size_t)TOK_LOCAL * cDI);
  W.Bz   = alloc((size_t)TOK_LOCAL * cDI);
  W.C    = alloc((size_t)TOK_LOCAL * cDI);
  W.dbc  = alloc((size_t)TOK_LOCAL * 48);
  W.yf   = alloc((size_t)TOK_LOCAL * cDM);
  W.node = alloc((size_t)NODES * cDM);
  W.srt  = alloc((size_t)NODES * cDM);
  W.ng   = alloc((size_t)NODES * cDM);
  W.mpx  = alloc((size_t)NODES * cDM);
  W.mpx2 = alloc((size_t)NODES * cDM);
  W.m    = alloc((size_t)NODES * cDM);
  W.cat  = alloc((size_t)NODES * 2 * cDM);
  W.rep  = alloc((size_t)NODES * cDM);
  W.ph   = alloc((size_t)NODES * cDM);
  float* z1    = alloc((size_t)NODES * 256);
  float* z2    = alloc((size_t)NODES * 256);
  float* zn1   = alloc((size_t)NODES * 256);
  float* zn2   = alloc((size_t)NODES * 256);
  float* part1 = alloc(NODES);
  float* part2 = alloc(NODES);

  // f16 weight arena (converted once per launch, on stream)
  _Float16* hbase = (_Float16*)(wsf + off);
  size_t hoff = 0;
  auto cvt = [&](const float* src, size_t n) -> const _Float16* {
    _Float16* dst = hbase + hoff; hoff += n;
    cvt_f16_kernel<<<DIV_UP((int)n, 256), 256, 0, stream>>>(src, dst, (int)n);
    return dst;
  };
  HalfWeights HW;
  HW.enc_w1 = cvt(enc_w1, (size_t)cMH * cHC);
  HW.enc_w2 = cvt(enc_w2, (size_t)cDM * cMH);
  glob.in_wh  = cvt(glob.in_w,  (size_t)4 * 2 * cDI * cDM);
  glob.xp_wh  = cvt(glob.xp_w,  (size_t)4 * 48 * cDI);
  glob.out_wh = cvt(glob.out_w, (size_t)4 * cDM * cDI);
  locl.in_wh  = cvt(locl.in_w,  (size_t)4 * 2 * cDI * cDM);
  locl.xp_wh  = cvt(locl.xp_w,  (size_t)4 * 48 * cDI);
  locl.out_wh = cvt(locl.out_w, (size_t)4 * cDM * cDI);
  HW.mw1 = cvt(mw1, (size_t)cML * cDM * cDM);
  HW.mw2 = cvt(mw2, (size_t)cML * cDM * cDM);
  HW.uw1 = cvt(uw1, (size_t)cML * cDM * 2 * cDM);
  HW.uw2 = cvt(uw2, (size_t)cML * cDM * cDM);
  HW.pw1 = cvt(pw1, (size_t)cDM * cDM);
  HW.pw2 = cvt(pw2, (size_t)256 * cDM);

  run_view(stream, pooled1, perm1, edges1, enc_b1, enc_rms, enc_b2,
           locl, glob, mb1, mb2, ub1, ub2, pb1, pb2, HW, z1, W);
  run_view(stream, pooled2, perm2, edges2, enc_b1, enc_rms, enc_b2,
           locl, glob, mb1, mb2, ub1, ub2, pb1, pb2, HW, z2, W);

  row_normalize_kernel<<<DIV_UP(NODES * 32, 256), 256, 0, stream>>>(z1, zn1, NODES);
  row_normalize_kernel<<<DIV_UP(NODES * 32, 256), 256, 0, stream>>>(z2, zn2, NODES);
  infonce_rows_kernel<<<NODES, 256, 0, stream>>>(zn1, zn2, part1);
  infonce_rows_kernel<<<NODES, 256, 0, stream>>>(zn2, zn1, part2);
  final_loss_kernel<<<1, 256, 0, stream>>>(part1, part2, (float*)d_out);
}
